// Head_40673340293910
// MI455X (gfx1250) — compile-verified, matrix operand order checked by reading, not run on hardware
//
#include <hip/hip_runtime.h>
#include <hip/hip_bf16.h>

#define BB 8
#define TT 2048
#define DD 1024
#define HH 64

typedef __attribute__((ext_vector_type(16))) __bf16 v16bf;
typedef __attribute__((ext_vector_type(8)))  __bf16 v8bf;
typedef __attribute__((ext_vector_type(8)))  float  v8f;
typedef __attribute__((ext_vector_type(4)))  float  v4f;

__device__ __forceinline__ v8f wmma_bf16(v16bf a, v16bf b, v8f c) {
    return __builtin_amdgcn_wmma_f32_16x16x32_bf16(false, a, false, b, (short)0, c,
                                                   false, false);
}
__device__ __forceinline__ v4f  ld4f(const float* p)  { return *(const v4f*)p; }
__device__ __forceinline__ v8bf ld8bf(const __bf16* p){ return *(const v8bf*)p; }
__device__ __forceinline__ v16bf cat(v8bf lo, v8bf hi) {
    return __builtin_shufflevector(lo, hi, 0,1,2,3,4,5,6,7,8,9,10,11,12,13,14,15);
}

// ---------------------------------------------------------------------------
// Kernel 0: transpose + convert weights: W (D x H fp32) -> wt[w][n][k] bf16.
// ---------------------------------------------------------------------------
__global__ void prep_w_kernel(const float* __restrict__ Wk,
                              const float* __restrict__ Wq,
                              const float* __restrict__ Wv,
                              __bf16* __restrict__ wt) {
    const int n = blockIdx.x % HH;
    const int w = blockIdx.x / HH;           // 0=q, 1=k, 2=v
    const float* W = (w == 0) ? Wq : (w == 1) ? Wk : Wv;
    __bf16* dst = wt + ((size_t)w * HH + n) * DD;
    for (int k = threadIdx.x; k < DD; k += blockDim.x)
        dst[k] = (__bf16)W[(size_t)k * HH + n];
}

// ---------------------------------------------------------------------------
// Kernel 1: QKV projection.  X (B*T x D) fp32 @ Wt^T -> q/k row-major bf16,
// v stored TRANSPOSED (vt[b][h][t]) so the AV pass reads contiguous columns.
// One wave computes a 16(M) x 64(N) strip.
// ---------------------------------------------------------------------------
__global__ __launch_bounds__(32)
void proj_kernel(const float* __restrict__ x,
                 const __bf16* __restrict__ wt,
                 __bf16* __restrict__ qb,
                 __bf16* __restrict__ kbuf,
                 __bf16* __restrict__ vt) {
    const int lane  = threadIdx.x;
    const int row0  = blockIdx.x * 16;           // row within B*T
    const int which = blockIdx.y;                // 0=q, 1=k, 2=v
    const int mrow  = row0 + (lane & 15);
    const int kb8   = (lane >> 4) << 3;          // 0 or 8
    const int n_lane = lane & 15;

    v8f acc[4] = {};
    for (int k0 = 0; k0 < DD; k0 += 32) {
        const float* xp = x + (size_t)mrow * DD + k0 + kb8;
        v4f x0 = ld4f(xp), x1 = ld4f(xp + 4), x2 = ld4f(xp + 16), x3 = ld4f(xp + 20);
        v16bf a;
        #pragma unroll
        for (int i = 0; i < 4; ++i) {
            a[i]      = (__bf16)x0[i];
            a[i + 4]  = (__bf16)x1[i];
            a[i + 8]  = (__bf16)x2[i];
            a[i + 12] = (__bf16)x3[i];
        }
        #pragma unroll
        for (int nt = 0; nt < 4; ++nt) {
            const __bf16* wrow = wt + ((size_t)which * HH + nt * 16 + n_lane) * DD;
            v16bf bm = cat(ld8bf(wrow + k0 + kb8), ld8bf(wrow + k0 + 16 + kb8));
            acc[nt] = wmma_bf16(a, bm, acc[nt]);
        }
    }
    const int mhi = (lane >> 4) * 8;
    if (which < 2) {
        __bf16* outp = (which == 0) ? qb : kbuf;
        #pragma unroll
        for (int nt = 0; nt < 4; ++nt)
            #pragma unroll
            for (int j = 0; j < 8; ++j)
                outp[(size_t)(row0 + mhi + j) * HH + nt * 16 + n_lane] =
                    (__bf16)acc[nt][j];
    } else {
        #pragma unroll
        for (int nt = 0; nt < 4; ++nt)
            #pragma unroll
            for (int j = 0; j < 8; ++j) {
                const int row   = row0 + mhi + j;        // in [0, B*T)
                const int batch = row >> 11;             // /T
                const int t     = row & (TT - 1);
                vt[((size_t)batch * HH + nt * 16 + n_lane) * TT + t] =
                    (__bf16)acc[nt][j];
            }
    }
}

// ---------------------------------------------------------------------------
// Kernel 2: scores tile = q . k^T * scale, causal mask -> fp32 S (T x T).
// ---------------------------------------------------------------------------
__global__ __launch_bounds__(32)
void scores_kernel(const __bf16* __restrict__ qb,
                   const __bf16* __restrict__ kbuf,
                   float* __restrict__ S, int batch_off) {
    const int lane   = threadIdx.x;
    const int q0     = blockIdx.x * 16;
    const int c0     = blockIdx.y * 16;
    const int n_lane = lane & 15;
    const int mhi    = (lane >> 4) * 8;
    const float NEG_INF = -__builtin_inff();

    if (c0 > q0 + 15) {                          // entirely above diagonal
        #pragma unroll
        for (int j = 0; j < 8; ++j)
            S[(size_t)(q0 + mhi + j) * TT + c0 + n_lane] = NEG_INF;
        return;
    }

    const int kb8 = (lane >> 4) << 3;
    const __bf16* qrow = qb   + batch_off + (size_t)(q0 + (lane & 15)) * HH;
    const __bf16* krow = kbuf + batch_off + (size_t)(c0 + (lane & 15)) * HH;

    v8f acc = {};
    #pragma unroll
    for (int h0 = 0; h0 < HH; h0 += 32) {
        v16bf a  = cat(ld8bf(qrow + h0 + kb8), ld8bf(qrow + h0 + 16 + kb8));
        v16bf bm = cat(ld8bf(krow + h0 + kb8), ld8bf(krow + h0 + 16 + kb8));
        acc = wmma_bf16(a, bm, acc);
    }

    const float scale = 0.125f;                  // 64^-0.5
    #pragma unroll
    for (int j = 0; j < 8; ++j) {
        const int m = q0 + mhi + j;
        const int c = c0 + n_lane;
        S[(size_t)m * TT + c] = (c <= m) ? acc[j] * scale : NEG_INF;
    }
}

// ---------------------------------------------------------------------------
// Kernel 3: per-column (reduce over QUERY axis) online max & sum-exp.
// Streams 16MB of S; prefetch a few rows ahead (global_prefetch_b8).
// ---------------------------------------------------------------------------
__global__ void colstats_kernel(const float* __restrict__ S,
                                float* __restrict__ colmax,
                                float* __restrict__ colrsum) {
    const int c = blockIdx.x * blockDim.x + threadIdx.x;
    const float NEG_INF = -__builtin_inff();
    float m = NEG_INF, l = 0.f;
    for (int q = 0; q < TT; ++q) {
        if (q + 8 < TT)
            __builtin_prefetch(S + (size_t)(q + 8) * TT + c, 0, 3);
        const float xv = S[(size_t)q * TT + c];
        if (xv > m) { l *= __expf(m - xv); m = xv; }
        if (xv > NEG_INF) l += __expf(xv - m);
    }
    colmax[c]  = m;
    colrsum[c] = 1.f / l;
}

// ---------------------------------------------------------------------------
// Kernel 4: out = attn @ v with attn[q,c] = exp(S[q,c]-colmax[c])*colrsum[c].
// One wave per 16-query strip produces ALL H=64 columns (4 accumulators):
// the S read + exp normalization (dominant traffic) happens once, not 4x.
// ---------------------------------------------------------------------------
__global__ __launch_bounds__(32)
void av_kernel(const float* __restrict__ S,
               const __bf16* __restrict__ vt, int batch_off,
               const float* __restrict__ colmax,
               const float* __restrict__ colrsum,
               float* __restrict__ out, int out_off) {
    const int lane   = threadIdx.x;
    const int q0     = blockIdx.x * 16;
    const int n_lane = lane & 15;
    const int kb8    = (lane >> 4) << 3;
    const int qrow   = q0 + (lane & 15);
    const float* srow = S + (size_t)qrow * TT;

    v8f acc[4] = {};
    for (int k0 = 0; k0 < q0 + 16; k0 += 32) {   // beyond-diagonal entries give 0
        const float* sp = srow    + k0 + kb8;
        const float* mp = colmax  + k0 + kb8;
        const float* rp = colrsum + k0 + kb8;
        v4f sA = ld4f(sp),      sB = ld4f(sp + 4),
            sC = ld4f(sp + 16), sD = ld4f(sp + 20);
        v4f mA = ld4f(mp),      mB = ld4f(mp + 4),
            mC = ld4f(mp + 16), mD = ld4f(mp + 20);
        v4f rA = ld4f(rp),      rB = ld4f(rp + 4),
            rC = ld4f(rp + 16), rD = ld4f(rp + 20);
        v16bf a;
        #pragma unroll
        for (int i = 0; i < 4; ++i) {
            a[i]      = (__bf16)(__expf(sA[i] - mA[i]) * rA[i]);
            a[i + 4]  = (__bf16)(__expf(sB[i] - mB[i]) * rB[i]);
            a[i + 8]  = (__bf16)(__expf(sC[i] - mC[i]) * rC[i]);
            a[i + 12] = (__bf16)(__expf(sD[i] - mD[i]) * rD[i]);
        }
        #pragma unroll
        for (int ht = 0; ht < 4; ++ht) {
            const __bf16* vrow =
                vt + batch_off + (size_t)(ht * 16 + n_lane) * TT;
            v16bf bm = cat(ld8bf(vrow + k0 + kb8), ld8bf(vrow + k0 + 16 + kb8));
            acc[ht] = wmma_bf16(a, bm, acc[ht]);
        }
    }

    const int mhi = (lane >> 4) * 8;
    #pragma unroll
    for (int ht = 0; ht < 4; ++ht)
        #pragma unroll
        for (int j = 0; j < 8; ++j)
            out[out_off + (size_t)(q0 + mhi + j) * HH + ht * 16 + n_lane] =
                acc[ht][j];
}

// ---------------------------------------------------------------------------
extern "C" void kernel_launch(void* const* d_in, const int* in_sizes, int n_in,
                              void* d_out, int out_size, void* d_ws, size_t ws_size,
                              hipStream_t stream) {
    const float* x  = (const float*)d_in[0];
    const float* Wk = (const float*)d_in[1];
    const float* Wq = (const float*)d_in[2];
    const float* Wv = (const float*)d_in[3];
    float* out = (float*)d_out;

    // Workspace layout (~22.4 MB):
    //   qb, kbuf : bf16 [B*T*H]                (2 x 2 MB)
    //   vt       : bf16 [B*H*T] (transposed)   (2 MB)
    //   wt       : bf16 [3*H*D] (transposed)   (384 KB)
    //   S        : fp32 [T*T] (reused per b)   (16 MB)
    //   colmax, colrsum : fp32 [T] each
    __bf16* qb   = (__bf16*)d_ws;
    __bf16* kbuf = qb   + (size_t)BB * TT * HH;
    __bf16* vt   = kbuf + (size_t)BB * TT * HH;
    __bf16* wt   = vt   + (size_t)BB * TT * HH;
    float*  S    = (float*)(wt + (size_t)3 * HH * DD);
    float*  colmax  = S + (size_t)TT * TT;
    float*  colrsum = colmax + TT;

    prep_w_kernel<<<3 * HH, 256, 0, stream>>>(Wk, Wq, Wv, wt);
    proj_kernel<<<dim3((BB * TT) / 16, 3), 32, 0, stream>>>(x, wt, qb, kbuf, vt);
    for (int b = 0; b < BB; ++b) {
        const int boff = b * TT * HH;
        scores_kernel<<<dim3(TT / 16, TT / 16), 32, 0, stream>>>(qb, kbuf, S, boff);
        colstats_kernel<<<TT / 256, 256, 0, stream>>>(S, colmax, colrsum);
        av_kernel<<<TT / 16, 32, 0, stream>>>(S, vt, boff, colmax, colrsum,
                                              out, boff);
    }
}
